// RNNDecoder_55181739819606
// MI455X (gfx1250) — matrix operand order repaired and checked by reading, not compiled, not run on hardware
//
#include <hip/hip_runtime.h>

// ---------------------------------------------------------------------------
// CDNA5 (gfx1250) LSTM decoder.  B=16, H=512, V=32000, L=2, T steps.
// f16 weights/activations, f32 accumulate via v_wmma_f32_16x16x32_f16.
// ---------------------------------------------------------------------------

typedef __attribute__((ext_vector_type(16))) _Float16 v16h;
typedef __attribute__((ext_vector_type(8)))  _Float16 v8h;
typedef __attribute__((ext_vector_type(8)))  float    v8f;

#define B_   16
#define H_   512
#define V_   32000
#define TILES_PER_WAVE 2
#define LOGITS_BLOCKS 250      // 250 blocks * 4 waves * 2 tiles = 2000 tiles
#define KBLOCKS 16             // 512 / 32
#define KB_PER_WAVE 4          // lstm split-K: 4 waves x 4 k-blocks

__device__ __forceinline__ v8f wmma16x16x32(v16h a, v16h b, v8f c) {
  return __builtin_amdgcn_wmma_f32_16x16x32_f16(
      /*neg_a=*/false, a, /*neg_b=*/false, b,
      /*c_mod=*/(short)0, c, /*reuse_a=*/false, /*reuse_b=*/false);
}

// A fragment (16x32, row-major source, ld = 512 elements).
// lane<16: row=lane,     K = {k0..k0+7, k0+16..k0+23}
// lane>=16: row=lane-16, K = {k0+8..k0+15, k0+24..k0+31}
__device__ __forceinline__ v16h load_a_frag(const _Float16* __restrict__ base,
                                            int k0, int lane) {
  int hf = lane >> 4, m = lane & 15;
  const _Float16* p = base + (size_t)m * H_ + k0 + hf * 8;
  union { v16h v; v8h h[2]; } u;
  u.h[0] = *(const v8h*)p;
  u.h[1] = *(const v8h*)(p + 16);
  return u.v;
}

// A fragment with per-row gather (embedding rows from W_tied, ld = 512).
__device__ __forceinline__ v16h load_a_gather(const _Float16* __restrict__ Wt,
                                              const int* __restrict__ rows,
                                              int k0, int lane) {
  int hf = lane >> 4, m = lane & 15;
  const _Float16* p = Wt + (size_t)rows[m] * H_ + k0 + hf * 8;
  union { v16h v; v8h h[2]; } u;
  u.h[0] = *(const v8h*)p;
  u.h[1] = *(const v8h*)(p + 16);
  return u.v;
}

// B fragment (32x16) for g = act @ W^T : B[k][n] = W[n0+n][k], W row-major NxK.
// lane: col n = lane&15 ; K = {k0+8h..k0+8h+7, k0+16+8h..k0+23+8h}, h = lane>>4.
__device__ __forceinline__ v16h load_b_frag(const _Float16* __restrict__ W,
                                            int n0, int k0, int lane) {
  int hf = lane >> 4, n = lane & 15;
  const _Float16* p = W + (size_t)(n0 + n) * H_ + k0 + hf * 8;
  union { v16h v; v8h h[2]; } u;
  u.h[0] = *(const v8h*)p;
  u.h[1] = *(const v8h*)(p + 16);
  return u.v;
}

__device__ __forceinline__ float sigmoidf_(float x) {
  return 1.0f / (1.0f + expf(-x));
}

__device__ __forceinline__ unsigned fkey(float v) {
  unsigned u = __float_as_uint(v);
  return (u & 0x80000000u) ? ~u : (u | 0x80000000u);
}
__device__ __forceinline__ unsigned long long packvi(float v, int n) {
  // larger value wins; on tie, smaller index wins (~n larger for smaller n)
  return ((unsigned long long)fkey(v) << 32) | (unsigned)(~(unsigned)n);
}

// ---------------------------------------------------------------------------
// fp32 -> f16 conversion (one-time)
// ---------------------------------------------------------------------------
__global__ __launch_bounds__(256) void cvt_f32_to_f16(
    const float* __restrict__ src, _Float16* __restrict__ dst, size_t n) {
  size_t stride = (size_t)gridDim.x * blockDim.x;
  for (size_t i = (size_t)blockIdx.x * blockDim.x + threadIdx.x; i < n; i += stride)
    dst[i] = (_Float16)src[i];
}

// ---------------------------------------------------------------------------
// One LSTM layer for one step.  grid = 32 blocks x 128 threads.
// Block owns one 16-wide j-block of H; its 4 waves split K (4 k-blocks each)
// and each computes partial tiles for all 4 gates (A fragment shared).
// Partials are summed through LDS, then all 128 threads do the gate math.
// layer 0, t>0: finishes argmax reduction from `partials`, gathers embedding.
// ---------------------------------------------------------------------------
__global__ __launch_bounds__(128) void lstm_layer_kernel(
    int l, int t,
    const _Float16* __restrict__ Wih_h,   // [L][4H][H]
    const _Float16* __restrict__ Whh_h,   // [L][4H][H]
    const _Float16* __restrict__ Wt_h,    // [V][H]
    const _Float16* __restrict__ x_h,     // [B][H]
    const float* __restrict__ b_ih,       // [L][4H]
    const float* __restrict__ b_hh,       // [L][4H]
    _Float16* __restrict__ h_state,       // [2][L][B][H]  (step parity)
    float* __restrict__ c_state,          // [L][B][H]
    const unsigned long long* __restrict__ partials) {  // [250][16]

  __shared__ float gl[4][4][16][16];      // [wave][gate][m][jj]  16 KB
  __shared__ unsigned long long red[16][8];
  __shared__ int sidx[16];

  const int lane = threadIdx.x & 31;
  const int wave = threadIdx.x >> 5;

  // ---- finish argmax over vocab partials, get embedding row per batch ----
  if (l == 0 && t > 0) {
    int rr = threadIdx.x & 15;
    int ch = threadIdx.x >> 4;            // 0..7
    unsigned long long best = 0ull;
    for (int b = ch; b < LOGITS_BLOCKS; b += 8) {
      unsigned long long p = partials[(size_t)b * 16 + rr];
      if (p > best) best = p;
    }
    red[rr][ch] = best;
    __syncthreads();
    if (threadIdx.x < 16) {
      unsigned long long bb = red[threadIdx.x][0];
      #pragma unroll
      for (int c = 1; c < 8; ++c)
        if (red[threadIdx.x][c] > bb) bb = red[threadIdx.x][c];
      sidx[threadIdx.x] = (int)(~(unsigned)(bb & 0xffffffffull));
    }
    __syncthreads();
  }

  const int j0 = blockIdx.x * 16;         // 32 blocks cover H = 512
  const int parity = t & 1;
  const _Float16* Wih_l = Wih_h + (size_t)l * 4 * H_ * H_;
  const _Float16* Whh_l = Whh_h + (size_t)l * 4 * H_ * H_;
  const _Float16* h_prev = h_state + ((size_t)(parity ^ 1) * 2 + l) * B_ * H_;
  const _Float16* h_in0  = h_state + ((size_t)parity * 2 + 0) * B_ * H_;
  _Float16* h_out        = h_state + ((size_t)parity * 2 + l) * B_ * H_;

  // ---- phase 1: split-K partial GEMM (this wave: k-blocks kbase..kbase+3) --
  v8f acc[4] = {v8f{}, v8f{}, v8f{}, v8f{}};
  const int kbase = wave * KB_PER_WAVE;

  #pragma unroll
  for (int kk = 0; kk < KB_PER_WAVE; ++kk) {
    int k0 = (kbase + kk) * 32;
    v16h a0;
    if (l == 0) {
      if (t == 0) a0 = load_a_frag(x_h, k0, lane);
      else        a0 = load_a_gather(Wt_h, sidx, k0, lane);
    } else {
      a0 = load_a_frag(h_in0, k0, lane);
    }
    v16h bi[4];
    #pragma unroll
    for (int G = 0; G < 4; ++G)
      bi[G] = load_b_frag(Wih_l, G * H_ + j0, k0, lane);
    #pragma unroll
    for (int G = 0; G < 4; ++G)
      acc[G] = wmma16x16x32(a0, bi[G], acc[G]);

    if (t > 0) {   // h_prev == 0 at t == 0
      v16h a1 = load_a_frag(h_prev, k0, lane);
      v16h bh[4];
      #pragma unroll
      for (int G = 0; G < 4; ++G)
        bh[G] = load_b_frag(Whh_l, G * H_ + j0, k0, lane);
      #pragma unroll
      for (int G = 0; G < 4; ++G)
        acc[G] = wmma16x16x32(a1, bh[G], acc[G]);
    }
  }

  // ---- stash partial tiles in LDS ----
  {
    int hf = lane >> 4, jj = lane & 15;
    #pragma unroll
    for (int G = 0; G < 4; ++G)
      #pragma unroll
      for (int r = 0; r < 8; ++r)
        gl[wave][G][r + 8 * hf][jj] = acc[G][r];
  }
  __syncthreads();

  // ---- phase 2: reduce 4 K-partials, gate math, state update --------------
  for (int e = threadIdx.x; e < 256; e += 128) {
    int m = e >> 4, jj = e & 15;
    int jcol = j0 + jj;
    float g4[4];
    #pragma unroll
    for (int G = 0; G < 4; ++G) {
      int n = l * 4 * H_ + G * H_ + jcol;
      g4[G] = gl[0][G][m][jj] + gl[1][G][m][jj] +
              gl[2][G][m][jj] + gl[3][G][m][jj] + b_ih[n] + b_hh[n];
    }
    size_t cix = ((size_t)l * B_ + m) * H_ + jcol;
    float cprev = (t == 0) ? 0.0f : c_state[cix];
    float cn = sigmoidf_(g4[1]) * cprev + sigmoidf_(g4[0]) * tanhf(g4[2]);
    float hn = sigmoidf_(g4[3]) * tanhf(cn);
    c_state[cix] = cn;
    h_out[(size_t)m * H_ + jcol] = (_Float16)hn;
  }
}

// ---------------------------------------------------------------------------
// logits = h1 @ W_tied^T + b_pred ; write out[b][t][:] ; per-block argmax
// partials.  grid = 250 blocks x 128 threads; each wave owns 2 consecutive
// 16x16 tiles; A/B fragments are software-pipelined one k-block ahead.
// ---------------------------------------------------------------------------
__global__ __launch_bounds__(128) void logits_kernel(
    int t, int T,
    const _Float16* __restrict__ Wt_h,   // [V][H]
    const _Float16* __restrict__ h1,     // [B][H] (parity resolved layer-1 h)
    const float* __restrict__ b_pred,    // [V]
    float* __restrict__ out,             // [B][T][V]
    unsigned long long* __restrict__ partials) {  // [250][16]

  __shared__ unsigned long long warr[4][16];

  const int lane = threadIdx.x & 31;
  const int wave = threadIdx.x >> 5;
  const int tile0 = (blockIdx.x * 4 + wave) * TILES_PER_WAVE;  // 0..1998
  const int n00 = tile0 * 16;
  const int n01 = n00 + 16;

  v8f acc0 = v8f{}, acc1 = v8f{};

  // software pipeline: prefetch next k-block while multiplying current one
  v16h a  = load_a_frag(h1, 0, lane);
  v16h b0 = load_b_frag(Wt_h, n00, 0, lane);
  v16h b1 = load_b_frag(Wt_h, n01, 0, lane);
  for (int kb = 0; kb < KBLOCKS; ++kb) {
    v16h an = a, bn0 = b0, bn1 = b1;
    if (kb + 1 < KBLOCKS) {
      int k1 = (kb + 1) * 32;
      an  = load_a_frag(h1, k1, lane);
      bn0 = load_b_frag(Wt_h, n00, k1, lane);
      bn1 = load_b_frag(Wt_h, n01, k1, lane);
    }
    acc0 = wmma16x16x32(a, b0, acc0);
    acc1 = wmma16x16x32(a, b1, acc1);
    a = an; b0 = bn0; b1 = bn1;
  }

  const int hf = lane >> 4;
  const float bp0 = b_pred[n00 + (lane & 15)];
  const float bp1 = b_pred[n01 + (lane & 15)];

  #pragma unroll
  for (int r = 0; r < 8; ++r) {
    int m = r + 8 * hf;
    int c0 = n00 + (lane & 15);
    int c1 = n01 + (lane & 15);
    float v0 = acc0[r] + bp0;
    float v1 = acc1[r] + bp1;
    size_t rowbase = ((size_t)m * T + t) * V_;
    out[rowbase + c0] = v0;
    out[rowbase + c1] = v1;

    float bv = v0; int bn = c0;
    if (v1 > bv) { bv = v1; bn = c1; }
    // per-row max across this wave's 16 lanes (stay inside 16-lane group)
    #pragma unroll
    for (int off = 8; off >= 1; off >>= 1) {
      float ov = __shfl_xor(bv, off, 32);
      int   on = __shfl_xor(bn, off, 32);
      if (ov > bv || (ov == bv && on < bn)) { bv = ov; bn = on; }
    }
    if ((lane & 15) == 0) warr[wave][m] = packvi(bv, bn);
  }
  __syncthreads();
  if (threadIdx.x < 16) {
    unsigned long long b0r = warr[0][threadIdx.x];
    #pragma unroll
    for (int w = 1; w < 4; ++w)
      if (warr[w][threadIdx.x] > b0r) b0r = warr[w][threadIdx.x];
    partials[(size_t)blockIdx.x * 16 + threadIdx.x] = b0r;
  }
}

// ---------------------------------------------------------------------------
extern "C" void kernel_launch(void* const* d_in, const int* in_sizes, int n_in,
                              void* d_out, int out_size, void* d_ws, size_t ws_size,
                              hipStream_t stream) {
  const float* x      = (const float*)d_in[0];   // [16][512]
  const float* Wt     = (const float*)d_in[1];   // [32000][512]
  const float* b_pred = (const float*)d_in[2];   // [32000]
  const float* Wih    = (const float*)d_in[3];   // [2][2048][512]
  const float* Whh    = (const float*)d_in[4];   // [2][2048][512]
  const float* bih    = (const float*)d_in[5];   // [2][2048]
  const float* bhh    = (const float*)d_in[6];   // [2][2048]
  float* out = (float*)d_out;

  const int T = out_size / (B_ * V_);            // = max_len (avoid D2H read)

  // workspace layout (all 16B-aligned offsets)
  char* p = (char*)d_ws;
  _Float16* Wt_h  = (_Float16*)p; p += (size_t)V_ * H_ * 2;         // 32.77 MB
  _Float16* Wih_h = (_Float16*)p; p += (size_t)2 * 4 * H_ * H_ * 2; // 4.19 MB
  _Float16* Whh_h = (_Float16*)p; p += (size_t)2 * 4 * H_ * H_ * 2; // 4.19 MB
  _Float16* x_h   = (_Float16*)p; p += (size_t)B_ * H_ * 2;
  _Float16* h_st  = (_Float16*)p; p += (size_t)2 * 2 * B_ * H_ * 2; // parity x L
  float*    c_st  = (float*)p;    p += (size_t)2 * B_ * H_ * 4;
  unsigned long long* partials = (unsigned long long*)p;             // 250 x 16

  // one-time weight conversion to f16
  cvt_f32_to_f16<<<2048, 256, 0, stream>>>(Wt,  Wt_h,  (size_t)V_ * H_);
  cvt_f32_to_f16<<<512,  256, 0, stream>>>(Wih, Wih_h, (size_t)2 * 4 * H_ * H_);
  cvt_f32_to_f16<<<512,  256, 0, stream>>>(Whh, Whh_h, (size_t)2 * 4 * H_ * H_);
  cvt_f32_to_f16<<<32,   256, 0, stream>>>(x,   x_h,   (size_t)B_ * H_);

  for (int t = 0; t < T; ++t) {
    lstm_layer_kernel<<<32, 128, 0, stream>>>(0, t, Wih_h, Whh_h, Wt_h, x_h,
                                              bih, bhh, h_st, c_st, partials);
    lstm_layer_kernel<<<32, 128, 0, stream>>>(1, t, Wih_h, Whh_h, Wt_h, x_h,
                                              bih, bhh, h_st, c_st, partials);
    const _Float16* h1 = h_st + ((size_t)(t & 1) * 2 + 1) * B_ * H_;
    logits_kernel<<<LOGITS_BLOCKS, 128, 0, stream>>>(t, T, Wt_h, h1, b_pred,
                                                     out, partials);
  }
}
